// AttentionDecoder_27049704030833
// MI455X (gfx1250) — compile-verified, matrix operand order checked by reading, not run on hardware
//
#include <hip/hip_runtime.h>
#include <math.h>

// ---------------- Types for WMMA ----------------
typedef __attribute__((ext_vector_type(16))) __bf16 v16bf;
typedef __attribute__((ext_vector_type(8)))  float  v8f;

union Frag {
    v16bf v;
    float4 q[2];
    __bf16 e[16];
};

// Fragment = elements [p..p+7] and [p+16..p+23] (16-bit A/B VGPR layout:
// lanes 0-15 hold K 0..7 & 16..23 of their row; lanes 16-31 pre-shift K by +8).
__device__ __forceinline__ v16bf load_frag_bf(const __bf16* p) {
    Frag f;
    f.q[0] = *(const float4*)(p);        // 8 x bf16 = 16B
    f.q[1] = *(const float4*)(p + 16);   // 8 x bf16 = 16B
    return f.v;
}
__device__ __forceinline__ v16bf load_frag_f32(const float* p) {
    Frag f;
#pragma unroll
    for (int j = 0; j < 8; ++j) f.e[j]     = (__bf16)p[j];
#pragma unroll
    for (int j = 0; j < 8; ++j) f.e[8 + j] = (__bf16)p[16 + j];
    return f.v;
}

// ---------------- Generic WMMA GEMM (templated, branch-free) ----------------
// D[m,n] = sum_k A[m,k] * W[n,k] (+bias[n] if HAS_BIAS)
// One wave per (16*MT) x 64 tile. Block order: tm fastest, so concurrently
// resident blocks share the same 64-col W stripe through L2.
// gridDim.y = split-K slices; slice s handles K range [s*Kslice, (s+1)*Kslice)
// and writes to D + s*d_slice_stride.
// All M, N, Kslice multiples of 16*MT, 64, 32 respectively at every call site.
template<int MT, bool A_F32, bool W_F32, bool HAS_BIAS>
__global__ __launch_bounds__(32)
void gemm_wmma(const void* __restrict__ Ap, int lda,
               const void* __restrict__ Wp, int ldw,
               const float* __restrict__ bias,
               float* __restrict__ D, long ldd, long d_slice_stride,
               int m_tiles, int Kslice)
{
    const int tile = blockIdx.x;
    const int tm = tile % m_tiles;        // tm fastest -> W-stripe L2 reuse
    const int tn = tile / m_tiles;
    const int lane = threadIdx.x & 31;
    const int half = lane >> 4;           // 0 or 1
    const int l15  = lane & 15;
    const int m0 = tm * (16 * MT);
    const int n0 = tn * 64;
    const size_t koff = (size_t)blockIdx.y * (size_t)Kslice;
    const int kb = half * 8;

    const float*  Af = (const float*)Ap;
    const __bf16* Ab = (const __bf16*)Ap;
    const float*  Wf = (const float*)Wp;
    const __bf16* Wb = (const __bf16*)Wp;

    D += (size_t)blockIdx.y * d_slice_stride;

    v8f acc[MT][4] = {};

    for (int k0 = 0; k0 < Kslice; k0 += 32) {
        // Issue all fragment loads first (one clause), then the WMMA burst;
        // counter-based waits let loads overlap the matrix pipe.
        v16bf a[MT];
#pragma unroll
        for (int mi = 0; mi < MT; ++mi) {
            const size_t off = (size_t)(m0 + mi * 16 + l15) * lda + koff + k0 + kb;
            if constexpr (A_F32) a[mi] = load_frag_f32(Af + off);
            else                 a[mi] = load_frag_bf (Ab + off);
        }
        v16bf b[4];
#pragma unroll
        for (int sub = 0; sub < 4; ++sub) {
            const size_t off = (size_t)(n0 + sub * 16 + l15) * ldw + koff + k0 + kb;
            if constexpr (W_F32) b[sub] = load_frag_f32(Wf + off);
            else                 b[sub] = load_frag_bf (Wb + off);
        }
#pragma unroll
        for (int sub = 0; sub < 4; ++sub)
#pragma unroll
            for (int mi = 0; mi < MT; ++mi)
                acc[mi][sub] = __builtin_amdgcn_wmma_f32_16x16x32_bf16(
                    false, a[mi], false, b[sub], (short)0, acc[mi][sub], false, false);
    }

    // C/D layout: VGPR r -> row base + r (lanes 0-15) / base + 8 + r (lanes 16-31)
#pragma unroll
    for (int mi = 0; mi < MT; ++mi) {
#pragma unroll
        for (int sub = 0; sub < 4; ++sub) {
            const int col = n0 + sub * 16 + l15;
            float bv = 0.0f;
            if constexpr (HAS_BIAS) bv = bias[col];
#pragma unroll
            for (int r = 0; r < 8; ++r) {
                const int row = m0 + mi * 16 + r + half * 8;
                D[(size_t)row * ldd + col] = acc[mi][sub][r] + bv;
            }
        }
    }
}

// ---------------- Problem dimensions ----------------
#define T_LEN 64
#define BATCH 32
#define SRC 128
#define EMB 512
#define HID 1024
#define ENC 2048
#define ATTN_D 256
#define VOCAB 32000
#define PRED_K 3584  // HID + ENC + EMB
#define GATES 4096
#define DP_SLICES 4      // split-K slices for dproj GEMM (K=1024 -> 256 each)
#define GT_SLICES 8      // split-K slices for gates GEMM (K=3072 -> 384 each)

// ---------------- Setup kernels ----------------
__global__ void embed_gather(const int* __restrict__ tgt, const float* __restrict__ embedding,
                             __bf16* __restrict__ emb_tb, __bf16* __restrict__ pred_in)
{
    const int idx = blockIdx.x * blockDim.x + threadIdx.x; // B*T*EMB
    const int e = idx & (EMB - 1);
    const int bt = idx >> 9;                  // b*T + t
    const int b = bt / T_LEN, t = bt % T_LEN;
    const float x = embedding[(size_t)tgt[bt] * EMB + e];
    emb_tb[((size_t)(t * BATCH + b)) * EMB + e] = (__bf16)x;
    pred_in[(size_t)bt * PRED_K + (HID + ENC) + e] = (__bf16)x;
}

__global__ void convert_sub(__bf16* __restrict__ dst, long dst_ld, long dst_off,
                            const float* __restrict__ src, long src_ld, long src_off,
                            int rows, int cols)
{
    const int idx = blockIdx.x * blockDim.x + threadIdx.x;
    if (idx >= rows * cols) return;
    const int r = idx / cols, c = idx % cols;
    dst[(size_t)r * dst_ld + dst_off + c] = (__bf16)src[(size_t)r * src_ld + src_off + c];
}

__global__ void convert_transpose(__bf16* __restrict__ dst, const float* __restrict__ src,
                                  int R, int C)
{
    const int idx = blockIdx.x * blockDim.x + threadIdx.x;
    if (idx >= R * C) return;
    const int r = idx / C, c = idx % C;
    dst[(size_t)c * R + r] = (__bf16)src[(size_t)r * C + c];
}

__global__ void bias_sum(const float* a, const float* b, float* out, int n)
{
    const int i = blockIdx.x * blockDim.x + threadIdx.x;
    if (i < n) out[i] = a[i] + b[i];
}

__global__ void init_state(const float* __restrict__ h0, const float* __restrict__ c0,
                           float* __restrict__ h, float* __restrict__ c,
                           __bf16* __restrict__ act)
{
    const int idx = blockIdx.x * blockDim.x + threadIdx.x; // B*HID
    h[idx] = h0[idx];
    c[idx] = c0[idx];
    const int b = idx >> 10, j = idx & (HID - 1);
    act[(size_t)b * (ENC + HID) + ENC + j] = (__bf16)h0[idx];
}

// ---------------- Per-step kernels ----------------
// Attention: block = one batch row, 128 threads. Reduces dproj split-K
// partials into LDS first (deterministic, no atomics).
__global__ __launch_bounds__(128)
void attn_step(const float* __restrict__ enc_proj,
               const float* __restrict__ dpart,   // [DP_SLICES][B][ATTN_D]
               const float* __restrict__ bd,
               const float* __restrict__ v, const float* __restrict__ enc_out,
               const unsigned char* __restrict__ mask,
               __bf16* __restrict__ act, __bf16* __restrict__ pred_in,
               float* __restrict__ attn_out, int t)
{
    const int b = blockIdx.x;
    const int s = threadIdx.x;
    __shared__ float sc[SRC];
    __shared__ float dpsh[ATTN_D];
    __shared__ float mred, sred;

    // reduce dproj partials + bias
    for (int a = s; a < ATTN_D; a += SRC) {
        float x = bd[a];
#pragma unroll
        for (int sl = 0; sl < DP_SLICES; ++sl)
            x += dpart[(size_t)sl * BATCH * ATTN_D + (size_t)b * ATTN_D + a];
        dpsh[a] = x;
    }
    __syncthreads();

    const float* ep = enc_proj + ((size_t)b * SRC + s) * ATTN_D;
    float e = 0.0f;
#pragma unroll 4
    for (int a = 0; a < ATTN_D; ++a) e += v[a] * tanhf(ep[a] + dpsh[a]);
    sc[s] = mask[b * SRC + s] ? e : -1e9f;
    __syncthreads();

    if (s == 0) {
        float m = sc[0];
        for (int i = 1; i < SRC; ++i) m = fmaxf(m, sc[i]);
        mred = m;
    }
    __syncthreads();
    sc[s] = __expf(sc[s] - mred);
    __syncthreads();
    if (s == 0) {
        float su = 0.0f;
        for (int i = 0; i < SRC; ++i) su += sc[i];
        sred = su;
    }
    __syncthreads();
    const float w = sc[s] / sred;
    sc[s] = w;
    attn_out[((size_t)b * T_LEN + t) * SRC + s] = w;
    __syncthreads();

    // context[e] = sum_s attn[s] * enc_out[b][s][e]
    const float* eb = enc_out + (size_t)b * SRC * ENC;
    for (int e2 = s; e2 < ENC; e2 += SRC) {
        float a2 = 0.0f;
        for (int ss = 0; ss < SRC; ++ss) a2 += sc[ss] * eb[(size_t)ss * ENC + e2];
        act[(size_t)b * (ENC + HID) + e2] = (__bf16)a2;
        pred_in[((size_t)b * T_LEN + t) * PRED_K + HID + e2] = (__bf16)a2;
    }
}

// LSTM pointwise: reduces gates split-K partials + precomputed x_gates[t]
// (which carries b_ih + b_hh). Gate order i, f, g, o.
__global__ void lstm_step(const float* __restrict__ gpart,   // [GT_SLICES][B][GATES]
                          const float* __restrict__ xg_t,    // [B][GATES]
                          float* __restrict__ h, float* __restrict__ c,
                          __bf16* __restrict__ act, __bf16* __restrict__ pred_in, int t)
{
    const int idx = blockIdx.x * blockDim.x + threadIdx.x; // B*HID
    const int b = idx >> 10, j = idx & (HID - 1);
    const size_t gb = (size_t)b * GATES;
    float gi = xg_t[gb + j];
    float gf = xg_t[gb + HID + j];
    float gg = xg_t[gb + 2 * HID + j];
    float go = xg_t[gb + 3 * HID + j];
#pragma unroll
    for (int sl = 0; sl < GT_SLICES; ++sl) {
        const float* g = gpart + (size_t)sl * BATCH * GATES + gb;
        gi += g[j];
        gf += g[HID + j];
        gg += g[2 * HID + j];
        go += g[3 * HID + j];
    }
    const float si = 1.0f / (1.0f + __expf(-gi));
    const float sf = 1.0f / (1.0f + __expf(-gf));
    const float so = 1.0f / (1.0f + __expf(-go));
    const float cn = sf * c[idx] + si * tanhf(gg);
    const float hn = so * tanhf(cn);
    c[idx] = cn;
    h[idx] = hn;
    act[(size_t)b * (ENC + HID) + ENC + j] = (__bf16)hn;
    pred_in[((size_t)b * T_LEN + t) * PRED_K + j] = (__bf16)hn;
}

__global__ void copy_hc(const float* __restrict__ h, const float* __restrict__ c,
                        float* __restrict__ oh, float* __restrict__ oc)
{
    const int idx = blockIdx.x * blockDim.x + threadIdx.x; // B*HID
    oh[idx] = h[idx];
    oc[idx] = c[idx];
}

// ---------------- Host launcher ----------------
extern "C" void kernel_launch(void* const* d_in, const int* in_sizes, int n_in,
                              void* d_out, int out_size, void* d_ws, size_t ws_size,
                              hipStream_t stream)
{
    const int*   tgt       = (const int*)d_in[0];
    const float* h0        = (const float*)d_in[1];
    const float* c0        = (const float*)d_in[2];
    const float* enc_out   = (const float*)d_in[3];
    const unsigned char* mask = (const unsigned char*)d_in[4];
    const float* embedding = (const float*)d_in[5];
    const float* Wd        = (const float*)d_in[6];
    const float* bd        = (const float*)d_in[7];
    const float* We        = (const float*)d_in[8];
    const float* be        = (const float*)d_in[9];
    const float* vvec      = (const float*)d_in[10];
    const float* W_ih      = (const float*)d_in[11];
    const float* W_hh      = (const float*)d_in[12];
    const float* b_ih      = (const float*)d_in[13];
    const float* b_hh      = (const float*)d_in[14];
    const float* W_out     = (const float*)d_in[15];
    const float* b_out     = (const float*)d_in[16];

    // Output layout: logits [B,T,V] | h [1,B,HID] | c [1,B,HID] | attn [B,T,S]
    float* out_logits = (float*)d_out;
    float* out_h      = out_logits + (size_t)BATCH * T_LEN * VOCAB;
    float* out_c      = out_h + BATCH * HID;
    float* out_attn   = out_c + BATCH * HID;

    // Workspace carve-up (256B aligned)
    char* wp = (char*)d_ws;
    auto carve = [&](size_t bytes) -> char* {
        char* p = wp;
        wp += (bytes + 255) & ~(size_t)255;
        return p;
    };
    __bf16* emb_tb   = (__bf16*)carve((size_t)T_LEN * BATCH * EMB * 2);    // [T*B, 512]
    __bf16* pred_in  = (__bf16*)carve((size_t)BATCH * T_LEN * PRED_K * 2); // [B*T, 3584]
    __bf16* We_t     = (__bf16*)carve((size_t)ATTN_D * ENC * 2);           // [256, 2048]
    __bf16* Wd_t     = (__bf16*)carve((size_t)ATTN_D * HID * 2);           // [256, 1024]
    __bf16* W_x      = (__bf16*)carve((size_t)GATES * EMB * 2);            // [4096, 512]
    __bf16* W_rec    = (__bf16*)carve((size_t)GATES * (ENC + HID) * 2);    // [4096, 3072]
    __bf16* act      = (__bf16*)carve((size_t)BATCH * (ENC + HID) * 2);    // [32, 3072] = [ctx ; h]
    float* enc_proj  = (float*)carve((size_t)BATCH * SRC * ATTN_D * 4);    // [32,128,256]
    float* x_gates   = (float*)carve((size_t)T_LEN * BATCH * GATES * 4);   // [T,B,4096]
    float* bsum      = (float*)carve((size_t)GATES * 4);
    float* h_st      = (float*)carve((size_t)BATCH * HID * 4);
    float* c_st      = (float*)carve((size_t)BATCH * HID * 4);
    float* dpart     = (float*)carve((size_t)DP_SLICES * BATCH * ATTN_D * 4);
    float* gpart     = (float*)carve((size_t)GT_SLICES * BATCH * GATES * 4);

    // Opportunistic bf16 copy of W_out (halves traffic of the dominant GEMM)
    const size_t wout_bytes = (size_t)VOCAB * PRED_K * 2;
    __bf16* W_out_bf = nullptr;
    {
        size_t used = (size_t)(wp - (char*)d_ws);
        if (used + wout_bytes <= ws_size) W_out_bf = (__bf16*)carve(wout_bytes);
    }

    // ---- One-time prep ----
    {
        int n = BATCH * T_LEN * EMB;
        embed_gather<<<n / 256, 256, 0, stream>>>(tgt, embedding, emb_tb, pred_in);
    }
    {
        int n = ENC * ATTN_D;
        convert_transpose<<<(n + 255) / 256, 256, 0, stream>>>(We_t, We, ENC, ATTN_D);
    }
    {
        int n = HID * ATTN_D;
        convert_transpose<<<(n + 255) / 256, 256, 0, stream>>>(Wd_t, Wd, HID, ATTN_D);
    }
    {   // W_x[n][k] = W_ih[n][k], k < 512
        int n = GATES * EMB;
        convert_sub<<<(n + 255) / 256, 256, 0, stream>>>(W_x, EMB, 0, W_ih, EMB + ENC, 0, GATES, EMB);
    }
    {   // W_rec[:, 0:2048] = W_ih[:, 512:2560]
        int n = GATES * ENC;
        convert_sub<<<(n + 255) / 256, 256, 0, stream>>>(W_rec, ENC + HID, 0, W_ih, EMB + ENC, EMB, GATES, ENC);
    }
    {   // W_rec[:, 2048:3072] = W_hh
        int n = GATES * HID;
        convert_sub<<<(n + 255) / 256, 256, 0, stream>>>(W_rec, ENC + HID, ENC, W_hh, HID, 0, GATES, HID);
    }
    if (W_out_bf) {
        int n = VOCAB * PRED_K;
        convert_sub<<<(n + 255) / 256, 256, 0, stream>>>(W_out_bf, PRED_K, 0, W_out, PRED_K, 0, VOCAB, PRED_K);
    }
    bias_sum<<<GATES / 256, 256, 0, stream>>>(b_ih, b_hh, bsum, GATES);
    init_state<<<(BATCH * HID) / 256, 256, 0, stream>>>(h0, c0, h_st, c_st, act);

    // enc_proj = enc_out @ We + be : M=4096, N=256, K=2048  (A fp32 on the fly)
    gemm_wmma<2, true, false, true><<<dim3((BATCH * SRC / 32) * (ATTN_D / 64), 1), 32, 0, stream>>>(
        enc_out, ENC, We_t, ENC, be,
        enc_proj, ATTN_D, 0, BATCH * SRC / 32, ENC);

    // x_gates = emb_tb @ W_x^T + (b_ih+b_hh) : M=2048, N=4096, K=512
    gemm_wmma<2, false, false, true><<<dim3((T_LEN * BATCH / 32) * (GATES / 64), 1), 32, 0, stream>>>(
        emb_tb, EMB, W_x, EMB, bsum,
        x_gates, GATES, 0, T_LEN * BATCH / 32, EMB);

    // ---- Recurrent loop over timesteps ----
    for (int t = 0; t < T_LEN; ++t) {
        // dproj partials: h @ Wd : M=32, N=256, K=1024 split 4 ways (bias added in attn)
        gemm_wmma<2, false, false, false><<<dim3(ATTN_D / 64, DP_SLICES), 32, 0, stream>>>(
            (const void*)(act + ENC), ENC + HID, Wd_t, HID, nullptr,
            dpart, ATTN_D, (long)BATCH * ATTN_D, 1, HID / DP_SLICES);

        // attention + context (reduces dproj partials, softmax, weighted sum)
        attn_step<<<BATCH, SRC, 0, stream>>>(enc_proj, dpart, bd, vvec, enc_out, mask,
                                             act, pred_in, out_attn, t);

        // gates partials: [ctx; h] @ W_rec^T : M=32, N=4096, K=3072 split 8 ways
        gemm_wmma<2, false, false, false><<<dim3(GATES / 64, GT_SLICES), 32, 0, stream>>>(
            act, ENC + HID, W_rec, ENC + HID, nullptr,
            gpart, GATES, (long)BATCH * GATES, 1, (ENC + HID) / GT_SLICES);

        // LSTM pointwise: reduce partials + x_gates[t], update h/c, stage pred_in
        lstm_step<<<(BATCH * HID) / 256, 256, 0, stream>>>(
            gpart, x_gates + (size_t)t * BATCH * GATES, h_st, c_st, act, pred_in, t);
    }

    // logits = pred_in @ W_out^T + b_out : M=2048, N=32000, K=3584
    // MT=4 (64x64 per wave), tm-fastest ordering for W-stripe L2 reuse.
    if (W_out_bf) {
        gemm_wmma<4, false, false, true><<<dim3((BATCH * T_LEN / 64) * (VOCAB / 64), 1), 32, 0, stream>>>(
            pred_in, PRED_K, W_out_bf, PRED_K, b_out,
            out_logits, VOCAB, 0, BATCH * T_LEN / 64, PRED_K);
    } else {
        gemm_wmma<4, false, true, true><<<dim3((BATCH * T_LEN / 64) * (VOCAB / 64), 1), 32, 0, stream>>>(
            pred_in, PRED_K, W_out, PRED_K, b_out,
            out_logits, VOCAB, 0, BATCH * T_LEN / 64, PRED_K);
    }

    copy_hc<<<(BATCH * HID) / 256, 256, 0, stream>>>(h_st, c_st, out_h, out_c);
}
